// NetGATSimple_43989055045725
// MI455X (gfx1250) — matrix operand order repaired and checked by reading, not compiled, not run on hardware
//
#include <hip/hip_runtime.h>
#include <hip/hip_bf16.h>
#include <math.h>

typedef __attribute__((ext_vector_type(16))) _Float16 v16h;
typedef __attribute__((ext_vector_type(8)))  _Float16 h8;
typedef __attribute__((ext_vector_type(8)))  float    v8f;

#define NEG_SLOPE 0.2f

// ---------------- helpers ----------------

// Monotone order-preserving float<->uint encoding for atomicMax on floats.
__device__ __forceinline__ unsigned enc_ord(float f) {
  unsigned u = __float_as_uint(f);
  return (u & 0x80000000u) ? ~u : (u | 0x80000000u);
}
__device__ __forceinline__ float dec_ord(unsigned u) {
  return __uint_as_float((u & 0x80000000u) ? (u & 0x7FFFFFFFu) : ~u);
}

__device__ __forceinline__ void edge_sd(const int* __restrict__ ei, int E, int e, int& s, int& d) {
  if (e < E) { s = ei[e]; d = ei[E + e]; }      // edge_index[0][e], edge_index[1][e]
  else       { s = d = e - E; }                  // appended self-loops
}

// ---------------- precision conversion ----------------

__global__ void f32_to_f16_kernel(const float* __restrict__ in, _Float16* __restrict__ out, size_t n) {
  size_t i = (size_t)blockIdx.x * blockDim.x + threadIdx.x;
  if (i < n) out[i] = (_Float16)in[i];
}

// Pack W[K x SRCN] (f32, row-major) into the exact per-lane B-fragment order:
// Bp[((t*2 + hi)*LDN + n)*16 + q] = W[(t*32 + hi*16 + q)*SRCN + n]   (zero-pad n >= SRCN)
// so a lane reads its 16 halves as two consecutive b128 loads.
template<int K, int LDN, int SRCN>
__global__ void pack_b_kernel(const float* __restrict__ W, _Float16* __restrict__ Bp) {
  int i = blockIdx.x * blockDim.x + threadIdx.x;
  if (i >= K * LDN) return;
  int q  = i & 15;
  int n  = (i >> 4) % LDN;
  int hi = (i / (16 * LDN)) & 1;
  int t0 = i / (32 * LDN);
  int k  = t0 * 32 + hi * 16 + q;
  float v = (n < SRCN) ? W[k * SRCN + n] : 0.0f;
  Bp[i] = (_Float16)v;
}

// ---------------- WMMA GEMM ----------------
// C[N x LDN] = A[N x K] (f16) * B[K x LDN] (pre-packed f16), f32 accumulate.
// blockDim.x = 32 * (LDN/16); wave w computes the 16x16 tile at columns [16w,16w+16).
// A fragment: k = 16*hi + s (contiguous, same order as packed B) -> two b128 loads.
template<int K, int LDN>
__global__ void gemm_f16_wmma(const _Float16* __restrict__ A, const _Float16* __restrict__ Bp,
                              float* __restrict__ C, int N) {
  const int wave = threadIdx.x >> 5;
  const int lane = threadIdx.x & 31;
  const int row0 = blockIdx.x << 4;
  const int m    = lane & 15;
  const int hi   = lane >> 4;

  int rowA = row0 + m; if (rowA >= N) rowA = N - 1;           // clamp (tail blocks)
  const _Float16* ar = A + (size_t)rowA * K + (hi << 4);      // + khiA
  const int ncol = (wave << 4) + m;
  const _Float16* bp = Bp + ((size_t)hi * LDN + ncol) * 16;

  v8f acc = {};
#pragma unroll
  for (int t = 0; t < K / 32; ++t) {
    h8 a0 = *(const h8*)(ar + t * 32);
    h8 a1 = *(const h8*)(ar + t * 32 + 8);
    const _Float16* bt = bp + (size_t)t * 2 * LDN * 16;
    h8 b0 = *(const h8*)(bt);
    h8 b1 = *(const h8*)(bt + 8);
    v16h a = __builtin_shufflevector(a0, a1, 0,1,2,3,4,5,6,7,8,9,10,11,12,13,14,15);
    v16h b = __builtin_shufflevector(b0, b1, 0,1,2,3,4,5,6,7,8,9,10,11,12,13,14,15);
    // (neg_a, A, neg_b, B, c_mod, C, reuse_a, reuse_b)
    acc = __builtin_amdgcn_wmma_f32_16x16x32_f16(false, a, false, b, (short)0, acc, false, false);
  }

  const int mo = hi << 3;                                      // D: lanes 16-31 hold rows M+8
  const int no = (wave << 4) + (lane & 15);
  float* cp = C + (size_t)(row0 + mo) * LDN + no;
  if (row0 + 15 < N) {
#pragma unroll
    for (int v = 0; v < 8; ++v) cp[v * LDN] = acc[v];
  } else {
#pragma unroll
    for (int v = 0; v < 8; ++v) if (row0 + mo + v < N) cp[v * LDN] = acc[v];
  }
}

// ---------------- attention logits: al[n,h] = sum_c h[n,h,c] * a[h,c] ----------------

__global__ void attn_logits_kernel(const float* __restrict__ hbuf, int hstride, int H, int C,
                                   const float* __restrict__ a_src, const float* __restrict__ a_dst,
                                   float* __restrict__ alsrc, float* __restrict__ aldst, int N) {
  int nid = blockIdx.x * blockDim.x + threadIdx.x;
  if (nid >= N) return;
  const float* hr = hbuf + (size_t)nid * hstride;
  for (int h = 0; h < H; ++h) {
    float s = 0.f, d = 0.f;
    for (int c = 0; c < C; ++c) {
      float v = hr[h * C + c];
      s = fmaf(v, a_src[h * C + c], s);
      d = fmaf(v, a_dst[h * C + c], d);
    }
    alsrc[(size_t)nid * H + h] = s;
    aldst[(size_t)nid * H + h] = d;
  }
}

// ---------------- segment softmax over edges ----------------

__global__ void edge_max_kernel(const float* __restrict__ alsrc, const float* __restrict__ aldst,
                                const int* __restrict__ ei, int E, int N, int H,
                                unsigned* __restrict__ menc) {
  int e = blockIdx.x * blockDim.x + threadIdx.x;
  if (e >= E + N) return;
  int s, d; edge_sd(ei, E, e, s, d);
  for (int h = 0; h < H; ++h) {
    float v = alsrc[(size_t)s * H + h] + aldst[(size_t)d * H + h];
    v = v > 0.f ? v : NEG_SLOPE * v;                 // leaky_relu
    atomicMax(&menc[(size_t)d * H + h], enc_ord(v));
  }
}

__global__ void edge_exp_kernel(const float* __restrict__ alsrc, const float* __restrict__ aldst,
                                const int* __restrict__ ei, int E, int N, int H,
                                const unsigned* __restrict__ menc,
                                float* __restrict__ eexp, float* __restrict__ denom) {
  int e = blockIdx.x * blockDim.x + threadIdx.x;
  if (e >= E + N) return;
  int s, d; edge_sd(ei, E, e, s, d);
  for (int h = 0; h < H; ++h) {
    float v = alsrc[(size_t)s * H + h] + aldst[(size_t)d * H + h];
    v = v > 0.f ? v : NEG_SLOPE * v;
    float ex = expf(v - dec_ord(menc[(size_t)d * H + h]));
    eexp[(size_t)e * H + h] = ex;
    atomicAdd(&denom[(size_t)d * H + h], ex);
  }
}

// One thread per (edge, head): alpha loaded once, h[src] gathered as float4 (b128),
// C scalar f32 atomics into agg[dst].
__global__ void edge_agg_kernel(const float* __restrict__ hbuf, int hstride,
                                const int* __restrict__ ei,
                                const float* __restrict__ eexp, const float* __restrict__ denom,
                                int E, int N, int H, int C, float* __restrict__ agg) {
  int t = blockIdx.x * blockDim.x + threadIdx.x;
  int total = (E + N) * H;
  if (t >= total) return;
  int e = t / H, h = t - e * H;
  int s, d; edge_sd(ei, E, e, s, d);
  float al = eexp[(size_t)e * H + h] / (denom[(size_t)d * H + h] + 1e-16f);
  const float4* hp = (const float4*)(hbuf + (size_t)s * hstride + h * C);
  float* ap = agg + (size_t)d * (H * C) + h * C;
  for (int c4 = 0; c4 < C / 4; ++c4) {
    float4 hv = hp[c4];
    atomicAdd(ap + 4 * c4 + 0, al * hv.x);
    atomicAdd(ap + 4 * c4 + 1, al * hv.y);
    atomicAdd(ap + 4 * c4 + 2, al * hv.z);
    atomicAdd(ap + 4 * c4 + 3, al * hv.w);
  }
}

// ---------------- layer-1 epilogue: bias + ELU, emit f16 for next GEMM ----------------

__global__ void bias_elu_f16_kernel(const float* __restrict__ agg1, const float* __restrict__ b1,
                                    _Float16* __restrict__ act1h, int N) {
  size_t i = (size_t)blockIdx.x * blockDim.x + threadIdx.x;
  if (i >= (size_t)N * 64) return;
  int j = (int)(i % 64);
  float v = agg1[i] + b1[j];
  v = v > 0.f ? v : (expf(v) - 1.0f);               // ELU
  act1h[i] = (_Float16)v;
}

// ---------------- final: + b2, log_softmax over 40 classes ----------------

__global__ void finish_kernel(const float* __restrict__ agg2, const float* __restrict__ b2,
                              float* __restrict__ out, int N) {
  int nid = blockIdx.x * blockDim.x + threadIdx.x;
  if (nid >= N) return;
  float v[40];
  float mx = -3.0e38f;
  for (int j = 0; j < 40; ++j) {
    v[j] = agg2[(size_t)nid * 40 + j] + b2[j];
    mx = fmaxf(mx, v[j]);
  }
  float s = 0.f;
  for (int j = 0; j < 40; ++j) s += expf(v[j] - mx);
  float ls = logf(s);
  for (int j = 0; j < 40; ++j) out[(size_t)nid * 40 + j] = v[j] - mx - ls;
}

// ---------------- host side ----------------

static inline unsigned cdiv(size_t a, unsigned b) { return (unsigned)((a + b - 1) / b); }

extern "C" void kernel_launch(void* const* d_in, const int* in_sizes, int n_in,
                              void* d_out, int out_size, void* d_ws, size_t ws_size,
                              hipStream_t stream) {
  (void)n_in; (void)out_size; (void)ws_size;

  const float* x   = (const float*)d_in[0];
  const int*   ei  = (const int*)d_in[1];   // edge_index [2,E]
  const float* W1  = (const float*)d_in[2];
  const float* as1 = (const float*)d_in[3];
  const float* ad1 = (const float*)d_in[4];
  const float* b1  = (const float*)d_in[5];
  const float* W2  = (const float*)d_in[6];
  const float* as2 = (const float*)d_in[7];
  const float* ad2 = (const float*)d_in[8];
  const float* b2  = (const float*)d_in[9];
  float* out = (float*)d_out;

  const int IN_CH = 256, HID = 16, HEADS = 4, OUT_CH = 40;
  const int HC1 = HEADS * HID;       // 64
  const int LD2 = 48;                // padded layer-2 width (3 WMMA tiles)
  const int N  = in_sizes[0] / IN_CH;
  const int E  = in_sizes[1] / 2;
  const int ET = E + N;

  // bump allocator over d_ws (~96 MB total)
  char* p = (char*)d_ws;
  auto alloc = [&](size_t bytes) -> void* {
    void* r = (void*)p;
    p += (bytes + 255) & ~(size_t)255;
    return r;
  };
  _Float16* xh     = (_Float16*)alloc((size_t)N * IN_CH * sizeof(_Float16));
  _Float16* w1p    = (_Float16*)alloc((size_t)IN_CH * HC1 * sizeof(_Float16));
  _Float16* w2p    = (_Float16*)alloc((size_t)HC1 * LD2 * sizeof(_Float16));
  float*    h1     = (float*)alloc((size_t)N * HC1 * sizeof(float));
  float*    alsrc1 = (float*)alloc((size_t)N * HEADS * sizeof(float));
  float*    aldst1 = (float*)alloc((size_t)N * HEADS * sizeof(float));
  unsigned* m1     = (unsigned*)alloc((size_t)N * HEADS * sizeof(unsigned));
  float*    den1   = (float*)alloc((size_t)N * HEADS * sizeof(float));
  float*    eexp1  = (float*)alloc((size_t)ET * HEADS * sizeof(float));
  float*    agg1   = (float*)alloc((size_t)N * HC1 * sizeof(float));
  _Float16* act1h  = (_Float16*)alloc((size_t)N * HC1 * sizeof(_Float16));
  float*    h2     = (float*)alloc((size_t)N * LD2 * sizeof(float));
  float*    alsrc2 = (float*)alloc((size_t)N * sizeof(float));
  float*    aldst2 = (float*)alloc((size_t)N * sizeof(float));
  unsigned* m2     = (unsigned*)alloc((size_t)N * sizeof(unsigned));
  float*    den2   = (float*)alloc((size_t)N * sizeof(float));
  float*    eexp2  = (float*)alloc((size_t)ET * sizeof(float));
  float*    agg2   = (float*)alloc((size_t)N * OUT_CH * sizeof(float));

  const unsigned TB = 256;

  // ---- precision prep / weight packing ----
  f32_to_f16_kernel<<<cdiv((size_t)N * IN_CH, TB), TB, 0, stream>>>(x, xh, (size_t)N * IN_CH);
  pack_b_kernel<256, 64, 64><<<cdiv((size_t)256 * 64, TB), TB, 0, stream>>>(W1, w1p);
  pack_b_kernel<64, 48, 40><<<cdiv((size_t)64 * 48, TB), TB, 0, stream>>>(W2, w2p);

  // ---- layer 1 ----
  gemm_f16_wmma<256, 64><<<cdiv((size_t)N, 16), 128, 0, stream>>>(xh, w1p, h1, N);
  attn_logits_kernel<<<cdiv((size_t)N, TB), TB, 0, stream>>>(h1, HC1, HEADS, HID,
                                                             as1, ad1, alsrc1, aldst1, N);
  hipMemsetAsync(m1,  0, (size_t)N * HEADS * sizeof(unsigned), stream);  // enc-space minimum
  hipMemsetAsync(den1, 0, (size_t)N * HEADS * sizeof(float), stream);
  hipMemsetAsync(agg1, 0, (size_t)N * HC1 * sizeof(float), stream);
  edge_max_kernel<<<cdiv((size_t)ET, TB), TB, 0, stream>>>(alsrc1, aldst1, ei, E, N, HEADS, m1);
  edge_exp_kernel<<<cdiv((size_t)ET, TB), TB, 0, stream>>>(alsrc1, aldst1, ei, E, N, HEADS,
                                                           m1, eexp1, den1);
  edge_agg_kernel<<<cdiv((size_t)ET * HEADS, TB), TB, 0, stream>>>(h1, HC1, ei, eexp1, den1,
                                                                   E, N, HEADS, HID, agg1);
  bias_elu_f16_kernel<<<cdiv((size_t)N * HC1, TB), TB, 0, stream>>>(agg1, b1, act1h, N);

  // ---- layer 2 ----
  gemm_f16_wmma<64, 48><<<cdiv((size_t)N, 16), 96, 0, stream>>>(act1h, w2p, h2, N);
  attn_logits_kernel<<<cdiv((size_t)N, TB), TB, 0, stream>>>(h2, LD2, 1, OUT_CH,
                                                             as2, ad2, alsrc2, aldst2, N);
  hipMemsetAsync(m2,  0, (size_t)N * sizeof(unsigned), stream);
  hipMemsetAsync(den2, 0, (size_t)N * sizeof(float), stream);
  hipMemsetAsync(agg2, 0, (size_t)N * OUT_CH * sizeof(float), stream);
  edge_max_kernel<<<cdiv((size_t)ET, TB), TB, 0, stream>>>(alsrc2, aldst2, ei, E, N, 1, m2);
  edge_exp_kernel<<<cdiv((size_t)ET, TB), TB, 0, stream>>>(alsrc2, aldst2, ei, E, N, 1,
                                                           m2, eexp2, den2);
  edge_agg_kernel<<<cdiv((size_t)ET, TB), TB, 0, stream>>>(h2, LD2, ei, eexp2, den2,
                                                           E, N, 1, OUT_CH, agg2);
  finish_kernel<<<cdiv((size_t)N, TB), TB, 0, stream>>>(agg2, b2, out, N);
}
